// KAN_LAYER_56384330662446
// MI455X (gfx1250) — compile-verified, matrix operand order checked by reading, not run on hardware
//
#include <hip/hip_runtime.h>
#include <hip/hip_bf16.h>
#include <cstdint>

// ---- problem constants (from reference) ----
#define IN_DIM   1024
#define OUT_DIM  1024
#define BATCH    4096
#define KORD     3
#define NSPL     11            // NUM + K spline basis functions
#define NGRID    15            // NUM+1 + 2*K grid points per input dim
#define KPACK    12            // 1 (silu base) + 11 (spline basis)
#define KDIM     (IN_DIM * KPACK)   // 12288

// ---- GEMM tiling ----
#define TM 128
#define TN 128
#define TK 32
#define KSTEPS (KDIM / TK)     // 384

typedef __attribute__((ext_vector_type(16))) __bf16 v16bf;
typedef __attribute__((ext_vector_type(8)))  __bf16 v8bf;
typedef __attribute__((ext_vector_type(8)))  float  v8f;

// ---- CDNA5 async copy (GLOBAL_LOAD_ASYNC_TO_LDS_B128, ASYNCcnt) ----
// Probe result: builtin exists, 4 args, params are generic `int4*` pointers.
#if defined(__gfx1250__) && __has_builtin(__builtin_amdgcn_global_load_async_to_lds_b128)
#define ASYNC_LDS 1
typedef int v4i_vs __attribute__((vector_size(4 * sizeof(int))));
#define ASYNC_PTR(p) ((v4i_vs*)(p))
#endif

__device__ __forceinline__ void wait_async0() {
#if defined(__gfx1250__)
#if __has_builtin(__builtin_amdgcn_s_wait_asynccnt)
    __builtin_amdgcn_s_wait_asynccnt(0);
#else
    asm volatile("s_wait_asynccnt 0x0" ::: "memory");
#endif
#endif
}

// ------------------------------------------------------------------
// Kernel 1: pack augmented weights, N-major (Wt[o][i*12 + j], bf16)
//   j==0   : mask*scale_base            (multiplies silu(x_i))
//   j==1+k : coef[i,o,k]*mask*scale_sp  (multiplies basis k)
// ------------------------------------------------------------------
__global__ void kan_pack_weights(const float* __restrict__ coef,
                                 const float* __restrict__ scale_base,
                                 const float* __restrict__ scale_sp,
                                 const float* __restrict__ mask,
                                 __bf16* __restrict__ Wt) {
    int idx = blockIdx.x * blockDim.x + threadIdx.x;   // over IN_DIM*OUT_DIM
    if (idx >= IN_DIM * OUT_DIM) return;
    int i = idx / OUT_DIM;
    int o = idx - i * OUT_DIM;
    float m  = mask[idx];
    float wb = m * scale_base[idx];
    float ws = m * scale_sp[idx];
    __bf16* dst = Wt + (size_t)o * KDIM + i * KPACK;
    dst[0] = (__bf16)wb;
    const float* c = coef + (size_t)idx * NSPL;
#pragma unroll
    for (int k = 0; k < NSPL; ++k) dst[1 + k] = (__bf16)(c[k] * ws);
}

// ------------------------------------------------------------------
// Kernel 2: pack augmented activations (Act[b][i*12 + j], bf16)
//   Cox–de Boor recursion (order 3) + silu base
// ------------------------------------------------------------------
__global__ void kan_pack_acts(const float* __restrict__ x,
                              const float* __restrict__ grid,
                              __bf16* __restrict__ Act) {
    int idx = blockIdx.x * blockDim.x + threadIdx.x;   // over BATCH*IN_DIM
    if (idx >= BATCH * IN_DIM) return;
    int b = idx / IN_DIM;
    int i = idx - b * IN_DIM;
    float xv = x[idx];

    float g[NGRID];
#pragma unroll
    for (int j = 0; j < NGRID; ++j) g[j] = grid[i * NGRID + j];

    float v[NGRID - 1];
#pragma unroll
    for (int j = 0; j < NGRID - 1; ++j)
        v[j] = (xv >= g[j] && xv < g[j + 1]) ? 1.0f : 0.0f;

#pragma unroll
    for (int p = 1; p <= KORD; ++p) {
#pragma unroll
        for (int j = 0; j < NGRID - 1; ++j) {
            if (j < NGRID - 1 - p) {
                float t = (xv - g[j]) / (g[j + p] - g[j]) * v[j]
                        + (g[j + p + 1] - xv) / (g[j + p + 1] - g[j + 1]) * v[j + 1];
                v[j] = (t != t) ? 0.0f : t;   // nan_to_num
            }
        }
    }

    float s = xv / (1.0f + __expf(-xv));      // silu
    __bf16* dst = Act + (size_t)b * KDIM + i * KPACK;
    dst[0] = (__bf16)s;
#pragma unroll
    for (int k = 0; k < NSPL; ++k) dst[1 + k] = (__bf16)v[k];
}

// ------------------------------------------------------------------
// Kernel 3: out[4096,1024] = Act[4096,12288] * Wt[1024,12288]^T
// bf16 WMMA 16x16x32, f32 accum. 256 threads = 8 waves (wave32).
// Workgroup tile 128x128; wave tile 64(M)x32(N) = 4x2 WMMA frags.
// Double-buffered LDS; tiles staged via GLOBAL_LOAD_ASYNC_TO_LDS_B128.
// ------------------------------------------------------------------
__global__ __launch_bounds__(256, 1)
void kan_gemm(const __bf16* __restrict__ Act,
              const __bf16* __restrict__ Wt,
              float* __restrict__ out) {
    __shared__ __align__(16) __bf16 As[2][TM * TK];   // 8 KB per buffer
    __shared__ __align__(16) __bf16 Bs[2][TN * TK];   // 8 KB per buffer

    const int tid   = threadIdx.x;
    const int lane  = tid & 31;
    const int wave  = tid >> 5;
    const int lrow  = lane & 15;
    const int khalf = (lane >> 4) * 8;          // 0 or 8 (bf16 A/B lane layout)

    const int blockM = blockIdx.x * TM;
    const int blockN = blockIdx.y * TN;
    const int waveM  = (wave & 1) * 64;
    const int waveN  = (wave >> 1) * 32;

    const __bf16* gA = Act + (size_t)blockM * KDIM;
    const __bf16* gB = Wt  + (size_t)blockN * KDIM;

    // copy plan: 8 KB tile = 512 x 16B chunks; 256 threads x 2 chunks each
    const int c0 = tid * 2;

    v8f acc[4][2];
#pragma unroll
    for (int mf = 0; mf < 4; ++mf)
#pragma unroll
        for (int nf = 0; nf < 2; ++nf) acc[mf][nf] = (v8f){};

    // Stage one (A,B) K-tile into LDS buffer `buf`.
    auto load_tile = [&](int buf, int k0) {
#pragma unroll
        for (int r = 0; r < 2; ++r) {
            int c   = c0 + r;
            int row = c >> 2;             // 0..127
            int col = (c & 3) * 8;        // 0,8,16,24 (halves)
#if defined(ASYNC_LDS)
            __builtin_amdgcn_global_load_async_to_lds_b128(
                ASYNC_PTR(&gA[(size_t)row * KDIM + k0 + col]),
                ASYNC_PTR(&As[buf][row * TK + col]), 0, 0);
            __builtin_amdgcn_global_load_async_to_lds_b128(
                ASYNC_PTR(&gB[(size_t)row * KDIM + k0 + col]),
                ASYNC_PTR(&Bs[buf][row * TK + col]), 0, 0);
#else
            *(v8bf*)&As[buf][row * TK + col] =
                *(const v8bf*)&gA[(size_t)row * KDIM + k0 + col];
            *(v8bf*)&Bs[buf][row * TK + col] =
                *(const v8bf*)&gB[(size_t)row * KDIM + k0 + col];
#endif
        }
    };

    load_tile(0, 0);
    wait_async0();
    __syncthreads();

    for (int kt = 0; kt < KSTEPS; ++kt) {
        const int buf = kt & 1;

        if (kt + 2 < KSTEPS) {            // warm L2 for the tile after next
            int row = c0 >> 2;
            __builtin_prefetch(&gA[(size_t)row * KDIM + (size_t)(kt + 2) * TK], 0, 1);
            __builtin_prefetch(&gB[(size_t)row * KDIM + (size_t)(kt + 2) * TK], 0, 1);
        }
        if (kt + 1 < KSTEPS) load_tile(buf ^ 1, (kt + 1) * TK);

        // A fragments: lane l holds row (l&15); K 0..7/16..23 (lanes 0-15),
        // K 8..15/24..31 (lanes 16-31) -> two 16B LDS reads per fragment.
        v16bf afr[4];
#pragma unroll
        for (int mf = 0; mf < 4; ++mf) {
            const __bf16* p = &As[buf][(waveM + mf * 16 + lrow) * TK];
            v8bf lo = *(const v8bf*)&p[khalf];
            v8bf hi = *(const v8bf*)&p[16 + khalf];
            afr[mf] = __builtin_shufflevector(lo, hi, 0,1,2,3,4,5,6,7,8,9,10,11,12,13,14,15);
        }
        v16bf bfr[2];
#pragma unroll
        for (int nf = 0; nf < 2; ++nf) {
            const __bf16* p = &Bs[buf][(waveN + nf * 16 + lrow) * TK];
            v8bf lo = *(const v8bf*)&p[khalf];
            v8bf hi = *(const v8bf*)&p[16 + khalf];
            bfr[nf] = __builtin_shufflevector(lo, hi, 0,1,2,3,4,5,6,7,8,9,10,11,12,13,14,15);
        }

#pragma unroll
        for (int mf = 0; mf < 4; ++mf)
#pragma unroll
            for (int nf = 0; nf < 2; ++nf)
                acc[mf][nf] = __builtin_amdgcn_wmma_f32_16x16x32_bf16(
                    false, afr[mf], false, bfr[nf],
                    (short)0, acc[mf][nf], false, false);

        // my async prefetch of buf^1 must have landed before anyone reads it,
        // and everyone must be done reading buf before next overwrite
        wait_async0();
        __syncthreads();
    }

    // D layout: VGPR r, lanes 0-15 -> M=r, lanes 16-31 -> M=8+r; N = lane&15
#pragma unroll
    for (int mf = 0; mf < 4; ++mf)
#pragma unroll
        for (int nf = 0; nf < 2; ++nf) {
            const int n     = blockN + waveN + nf * 16 + lrow;
            const int mbase = blockM + waveM + mf * 16 + (lane >> 4) * 8;
#pragma unroll
            for (int r = 0; r < 8; ++r)
                out[(size_t)(mbase + r) * OUT_DIM + n] = acc[mf][nf][r];
        }
}

// ------------------------------------------------------------------
// Host launcher
// ------------------------------------------------------------------
extern "C" void kernel_launch(void* const* d_in, const int* in_sizes, int n_in,
                              void* d_out, int out_size, void* d_ws, size_t ws_size,
                              hipStream_t stream) {
    const float* x          = (const float*)d_in[0];
    const float* grid       = (const float*)d_in[1];
    const float* coef       = (const float*)d_in[2];
    const float* scale_base = (const float*)d_in[3];
    const float* scale_sp   = (const float*)d_in[4];
    const float* mask       = (const float*)d_in[5];
    float* out = (float*)d_out;

    // workspace: Act (4096x12288 bf16 = 96 MiB) + Wt (1024x12288 bf16 = 24 MiB)
    __bf16* Act = (__bf16*)d_ws;
    __bf16* Wt  = (__bf16*)((char*)d_ws + (size_t)BATCH * KDIM * sizeof(__bf16));

    kan_pack_weights<<<(IN_DIM * OUT_DIM) / 256, 256, 0, stream>>>(
        coef, scale_base, scale_sp, mask, Wt);
    kan_pack_acts<<<(BATCH * IN_DIM) / 256, 256, 0, stream>>>(x, grid, Act);

    dim3 g(BATCH / TM, OUT_DIM / TN);   // 32 x 8 workgroups
    kan_gemm<<<g, 256, 0, stream>>>(Act, Wt, out);
}
